// CrossAttentionPro_85641647882453
// MI455X (gfx1250) — matrix operand order repaired and checked by reading, not compiled, not run on hardware
//
#include <hip/hip_runtime.h>
#include <hip/hip_bf16.h>

typedef __attribute__((ext_vector_type(16))) __bf16 v16bf;
typedef __attribute__((ext_vector_type(8)))  float  v8f;
typedef unsigned short u16;
typedef unsigned int   u32;

// ---------- helpers ----------

__device__ __forceinline__ u16 f2bf(float f) {
  u32 u = __float_as_uint(f);
  u32 r = u + 0x7FFFu + ((u >> 16) & 1u);   // round-to-nearest-even
  return (u16)(r >> 16);
}

__device__ __forceinline__ float bf2f(u16 h) {
  return __uint_as_float(((u32)h) << 16);
}

// A-operand (16xK tile, K-step 32), 16-bit A layout per CDNA5 ISA 7.12.2:
// lane<16: row=lane, v0..3 hold K={0,1},{2,3},{4,5},{6,7}; v4..7 hold K=16..23
// lane>=16: same rows, K offset +8 (v0..3 -> K=8..15, v4..7 -> K=24..31)
__device__ __forceinline__ v16bf load_a_bf(const u16* base, int ld, int k0) {
  int lane = threadIdx.x & 31;
  int row = lane & 15;
  int hi  = (lane >> 4) & 1;
  const u16* p = base + (size_t)row * ld + k0 + hi * 8;
  union { v16bf v; u32 u[8]; } r;
#pragma unroll
  for (int i = 0; i < 4; ++i) r.u[i]     = *(const u32*)(p + 2 * i);
#pragma unroll
  for (int i = 0; i < 4; ++i) r.u[4 + i] = *(const u32*)(p + 16 + 2 * i);
  return r.v;
}

// B-operand (Kx16 tile, K-step 32) fed from row-major B^T ([16 cols, K]):
// dense 16-bit B layout (same K order as A): lanes 0-15 -> v0..3 K=0..7,
// v4..7 K=16..23; lanes 16-31 -> v0..3 K=8..15, v4..7 K=24..31.
__device__ __forceinline__ v16bf load_b_bf(const u16* base, int ld, int k0) {
  int lane = threadIdx.x & 31;
  int col = lane & 15;
  int hi  = (lane >> 4) & 1;
  const u16* p = base + (size_t)col * ld + k0 + hi * 8;
  union { v16bf v; u32 u[8]; } r;
#pragma unroll
  for (int i = 0; i < 4; ++i) r.u[i]     = *(const u32*)(p + 2 * i);
#pragma unroll
  for (int i = 0; i < 4; ++i) r.u[4 + i] = *(const u32*)(p + 16 + 2 * i);
  return r.v;
}

__device__ __forceinline__ v8f wmma_bf(v16bf a, v16bf b, v8f c) {
  return __builtin_amdgcn_wmma_f32_16x16x32_bf16(false, a, false, b, (short)0, c,
                                                 false, false);
}

// Async global->LDS copy, 16B per lane (GLOBAL_LOAD_ASYNC_TO_LDS_B128,
// tracked by ASYNCcnt). Low 32 bits of a generic LDS pointer are the LDS
// byte offset.
__device__ __forceinline__ void async_load_b128(const void* gsrc, void* lds_dst) {
  u32 l = (u32)(size_t)lds_dst;
  asm volatile("global_load_async_to_lds_b128 %0, %1, off"
               :: "v"(l), "v"(gsrc) : "memory");
}

__device__ __forceinline__ void wait_asynccnt0() {
  asm volatile("s_wait_asynccnt 0x0" ::: "memory");
}

// C/D layout: vgpr j, lane L -> (row = j + 8*(L>>4), col = L&15)

// ---------- kernels ----------

__global__ void cvt_bf16_kernel(const float* __restrict__ src, u16* __restrict__ dst, int n) {
  int i = blockIdx.x * blockDim.x + threadIdx.x;
  if (i < n) dst[i] = f2bf(src[i]);
}

// out = A @ W^T + bias, split into per-head q/k and transposed vT.
// A: [nRows, 512] bf16, W: [1536, 512] bf16. One wave -> 16x64 tile.
__global__ __launch_bounds__(32) void qkv_gemm_kernel(
    const u16* __restrict__ Abf, const u16* __restrict__ Wbf,
    const float* __restrict__ bias, int nRows, int rowsPerBatch,
    u16* __restrict__ q, u16* __restrict__ k, u16* __restrict__ vT)
{
  const int Ck = 512, Dh = 64, Hh = 8;
  int r0 = blockIdx.x * 16;
  int o0 = blockIdx.y * 64;
  int lane = threadIdx.x & 31;
  v8f acc[4] = {};
  const u16* Abase = Abf + (size_t)r0 * Ck;
  for (int k0 = 0; k0 < Ck; k0 += 32) {
    v16bf a = load_a_bf(Abase, Ck, k0);
#pragma unroll
    for (int n = 0; n < 4; ++n) {
      v16bf b = load_b_bf(Wbf + (size_t)(o0 + 16 * n) * Ck, Ck, k0);
      acc[n] = wmma_bf(a, b, acc[n]);
    }
  }
  int hi = lane >> 4, col = lane & 15;
#pragma unroll
  for (int n = 0; n < 4; ++n) {
    int o = o0 + 16 * n + col;
    int sec = o >> 9;          // 0:q 1:k 2:v
    int oc = o & 511;
    int h = oc >> 6, d = oc & 63;
    float bval = bias[o];
#pragma unroll
    for (int j = 0; j < 8; ++j) {
      int g = r0 + j + 8 * hi;
      int bb = g / rowsPerBatch, t = g % rowsPerBatch;
      u16 bv = f2bf(acc[n][j] + bval);
      size_t hb = (size_t)bb * Hh + h;
      if (sec == 0)      q[(hb * rowsPerBatch + t) * Dh + d] = bv;
      else if (sec == 1) k[(hb * rowsPerBatch + t) * Dh + d] = bv;
      else               vT[(hb * Dh + d) * rowsPerBatch + t] = bv;
    }
  }
}

// S[bh, t, m] = bf16( scale * sum_d A[bh,t,d]*Bm[bh,m,d] )   (NT, K=64)
__global__ __launch_bounds__(32) void score_gemm_kernel(
    const u16* __restrict__ A, const u16* __restrict__ Bm, u16* __restrict__ S,
    int T_, int M_, float scale)
{
  int bh = blockIdx.z;
  int t0 = blockIdx.x * 16, m0 = blockIdx.y * 64;
  int lane = threadIdx.x & 31;
  const u16* Ab = A + ((size_t)bh * T_ + t0) * 64;
  v8f acc[4] = {};
  for (int k0 = 0; k0 < 64; k0 += 32) {
    v16bf a = load_a_bf(Ab, 64, k0);
#pragma unroll
    for (int n = 0; n < 4; ++n) {
      v16bf b = load_b_bf(Bm + ((size_t)bh * M_ + m0 + 16 * n) * 64, 64, k0);
      acc[n] = wmma_bf(a, b, acc[n]);
    }
  }
  int hi = lane >> 4, col = lane & 15;
#pragma unroll
  for (int n = 0; n < 4; ++n)
#pragma unroll
    for (int j = 0; j < 8; ++j) {
      int t = t0 + j + 8 * hi;
      int m = m0 + 16 * n + col;
      S[((size_t)bh * T_ + t) * M_ + m] = f2bf(acc[n][j] * scale);
    }
}

// row-wise softmax over last dim (bf16 in/out, f32 math), one 256-thread block/row
__global__ void softmax_kernel(const u16* __restrict__ S, u16* __restrict__ P, int M_) {
  __shared__ float red[8];
  int row = blockIdx.x;
  const u16* src = S + (size_t)row * M_;
  u16* dst = P + (size_t)row * M_;
  int tid = threadIdx.x;
  float mx = -1e30f;
  for (int i = tid; i < M_; i += blockDim.x) mx = fmaxf(mx, bf2f(src[i]));
#pragma unroll
  for (int o = 16; o > 0; o >>= 1) mx = fmaxf(mx, __shfl_xor(mx, o));
  if ((tid & 31) == 0) red[tid >> 5] = mx;
  __syncthreads();
  float rm = -1e30f;
  for (int w = 0; w < (int)(blockDim.x >> 5); ++w) rm = fmaxf(rm, red[w]);
  __syncthreads();
  float s = 0.f;
  for (int i = tid; i < M_; i += blockDim.x) s += __expf(bf2f(src[i]) - rm);
#pragma unroll
  for (int o = 16; o > 0; o >>= 1) s += __shfl_xor(s, o);
  if ((tid & 31) == 0) red[tid >> 5] = s;
  __syncthreads();
  float tot = 0.f;
  for (int w = 0; w < (int)(blockDim.x >> 5); ++w) tot += red[w];
  float inv = 1.0f / tot;
  for (int i = tid; i < M_; i += blockDim.x)
    dst[i] = f2bf(__expf(bf2f(src[i]) - rm) * inv);
}

// cval[b, t, h*64 + d] = sum_m P[bh,t,m] * Vt[bh,d,m]   (NT, K=M)
__global__ __launch_bounds__(32) void pv_gemm_kernel(
    const u16* __restrict__ P, const u16* __restrict__ Vt, float* __restrict__ out,
    int T_, int M_, int H_, int C_)
{
  int bh = blockIdx.z;
  int b = bh / H_, h = bh % H_;
  int t0 = blockIdx.x * 16;
  int lane = threadIdx.x & 31;
  const u16* Ab = P + ((size_t)bh * T_ + t0) * M_;
  v8f acc[4] = {};
  for (int k0 = 0; k0 < M_; k0 += 32) {
    v16bf a = load_a_bf(Ab, M_, k0);
#pragma unroll
    for (int n = 0; n < 4; ++n) {
      v16bf bv = load_b_bf(Vt + ((size_t)bh * 64 + 16 * n) * M_, M_, k0);
      acc[n] = wmma_bf(a, bv, acc[n]);
    }
  }
  int hi = lane >> 4, col = lane & 15;
#pragma unroll
  for (int n = 0; n < 4; ++n)
#pragma unroll
    for (int j = 0; j < 8; ++j) {
      int t = t0 + j + 8 * hi;
      out[((size_t)b * T_ + t) * C_ + h * 64 + 16 * n + col] = acc[n][j];
    }
}

// Fused chained attention: chained = scale * S1 @ G^T (G stored [s,m]); mask;
// online softmax; O += P @ v_x via vT_x.
// 4 waves / workgroup: wave w owns 16-row A strip (LDS, async-staged); all
// waves share the 32-row G s-block, async-staged into a double-buffered LDS
// slab (ASYNCcnt-tracked DMA overlaps WMMA on the previous block).
__global__ __launch_bounds__(128) void chained_attn_kernel(
    const u16* __restrict__ S1, const u16* __restrict__ Gm,
    const u16* __restrict__ Vt, const int* __restrict__ mask,
    float scale, float* __restrict__ out, int T_, int M_, int H_, int C_)
{
  __shared__ __align__(16) u16 As[4][16 * 1024];   // 128 KB: per-wave A strips
  __shared__ __align__(16) u16 Bs[2][32 * 1024];   // 128 KB: shared G double-buffer
  __shared__ __align__(16) u16 Ps[4][16 * 32];     //   4 KB: per-wave P tiles
  int bh = blockIdx.z;
  int b = bh / H_, h = bh % H_;
  int wv = (threadIdx.x >> 5) & 3;
  int lane = threadIdx.x & 31;
  int t0 = blockIdx.x * 64 + wv * 16;

  // async-stage this wave's 16 x M A strip (raw x2y scores)
  const u16* Sb = S1 + ((size_t)bh * T_ + t0) * M_;
  int nvecA = 16 * M_ / 8;
  for (int i = lane; i < nvecA; i += 32)
    async_load_b128(Sb + i * 8, &As[wv][i * 8]);

  // prefetch first shared G s-block (32 x M), whole workgroup cooperates
  const u16* Gb = Gm + (size_t)bh * T_ * M_;
  int nvecB = 32 * M_ / 8;
  for (int i = threadIdx.x; i < nvecB; i += 128)
    async_load_b128(Gb + i * 8, &Bs[0][i * 8]);

  int hi = lane >> 4, col = lane & 15;
  float m_run[8], l_run[8];
  v8f accO[4] = {};
#pragma unroll
  for (int j = 0; j < 8; ++j) { m_run[j] = -1e30f; l_run[j] = 0.f; }

  int cur = 0;
  for (int s0 = 0; s0 < T_; s0 += 32) {
    wait_asynccnt0();     // my async issues done
    __syncthreads();      // everyone's done; prev block's LDS reads complete
    if (s0 + 32 < T_) {   // kick DMA for next s-block into the other buffer
      const u16* src = Gb + (size_t)(s0 + 32) * M_;
      for (int i = threadIdx.x; i < nvecB; i += 128)
        async_load_b128(src + i * 8, &Bs[cur ^ 1][i * 8]);
    }

    v8f sc0 = {}, sc1 = {};
    for (int k0 = 0; k0 < M_; k0 += 32) {
      v16bf a  = load_a_bf(As[wv], M_, k0);
      v16bf b0 = load_b_bf(&Bs[cur][0], M_, k0);
      v16bf b1 = load_b_bf(&Bs[cur][16 * M_], M_, k0);
      sc0 = wmma_bf(a, b0, sc0);
      sc1 = wmma_bf(a, b1, sc1);
    }

    float p0[8], p1[8];
#pragma unroll
    for (int j = 0; j < 8; ++j) {
      int t = t0 + j + 8 * hi;
      int s = s0 + col;
      float x0 = sc0[j] * scale;
      float x1 = sc1[j] * scale;
      if (mask[(size_t)t * T_ + s] == 0)      x0 = -1e30f;
      if (mask[(size_t)t * T_ + s + 16] == 0) x1 = -1e30f;
      p0[j] = x0; p1[j] = x1;
    }
#pragma unroll
    for (int j = 0; j < 8; ++j) {
      float v = fmaxf(p0[j], p1[j]);
      v = fmaxf(v, __shfl_xor(v, 8));
      v = fmaxf(v, __shfl_xor(v, 4));
      v = fmaxf(v, __shfl_xor(v, 2));
      v = fmaxf(v, __shfl_xor(v, 1));
      float mn = fmaxf(m_run[j], v);
      float al = __expf(m_run[j] - mn);
      m_run[j] = mn;
      float e0 = __expf(p0[j] - mn);
      float e1 = __expf(p1[j] - mn);
      Ps[wv][(j + 8 * hi) * 32 + col] = f2bf(e0);
      Ps[wv][(j + 8 * hi) * 32 + 16 + col] = f2bf(e1);
      float rs = e0 + e1;
      rs += __shfl_xor(rs, 8);
      rs += __shfl_xor(rs, 4);
      rs += __shfl_xor(rs, 2);
      rs += __shfl_xor(rs, 1);
      l_run[j] = l_run[j] * al + rs;
#pragma unroll
      for (int nt = 0; nt < 4; ++nt) accO[nt][j] *= al;
    }
    // P tile write->read is same-wave LDS (in-order per wave): no barrier needed
    v16bf ap = load_a_bf(Ps[wv], 32, 0);
#pragma unroll
    for (int nt = 0; nt < 4; ++nt) {
      v16bf bv = load_b_bf(Vt + ((size_t)bh * 64 + 16 * nt) * T_, T_, s0);
      accO[nt] = wmma_bf(ap, bv, accO[nt]);
    }
    cur ^= 1;
  }
#pragma unroll
  for (int nt = 0; nt < 4; ++nt)
#pragma unroll
    for (int j = 0; j < 8; ++j) {
      int t = t0 + j + 8 * hi;
      out[((size_t)b * T_ + t) * C_ + h * 64 + 16 * nt + col] = accO[nt][j] / l_run[j];
    }
}

__global__ void diff_bf_kernel(const float* __restrict__ a, const float* __restrict__ b,
                               u16* __restrict__ d, int n) {
  int i = blockIdx.x * blockDim.x + threadIdx.x;
  if (i < n) d[i] = f2bf(a[i] - b[i]);
}

// out = A @ W^T + bias, f32 out. A: [nRows, 512] bf16, W: [512, 512] bf16.
__global__ __launch_bounds__(32) void proj_gemm_kernel(
    const u16* __restrict__ Abf, const u16* __restrict__ Wbf,
    const float* __restrict__ bias, float* __restrict__ out, int nRows)
{
  const int Ck = 512;
  int r0 = blockIdx.x * 16;
  int o0 = blockIdx.y * 64;
  int lane = threadIdx.x & 31;
  v8f acc[4] = {};
  const u16* Abase = Abf + (size_t)r0 * Ck;
  for (int k0 = 0; k0 < Ck; k0 += 32) {
    v16bf a = load_a_bf(Abase, Ck, k0);
#pragma unroll
    for (int n = 0; n < 4; ++n) {
      v16bf b = load_b_bf(Wbf + (size_t)(o0 + 16 * n) * Ck, Ck, k0);
      acc[n] = wmma_bf(a, b, acc[n]);
    }
  }
  int hi = lane >> 4, col = lane & 15;
#pragma unroll
  for (int n = 0; n < 4; ++n) {
    int o = o0 + 16 * n + col;
    float bval = bias[o];
#pragma unroll
    for (int j = 0; j < 8; ++j) {
      int g = r0 + j + 8 * hi;
      out[(size_t)g * Ck + o] = acc[n][j] + bval;
    }
  }
}

// ---------- launcher ----------

extern "C" void kernel_launch(void* const* d_in, const int* in_sizes, int n_in,
                              void* d_out, int out_size, void* d_ws, size_t ws_size,
                              hipStream_t stream) {
  constexpr int B = 2, T = 2048, M = 1024, C = 512, H = 8, D = 64;
  constexpr int BH = B * H;
  const float scale = 0.125f;  // 1/sqrt(64)

  const float* x      = (const float*)d_in[0];
  const float* y      = (const float*)d_in[1];
  const int*   mask   = (const int*)d_in[2];
  const float* qkv_w  = (const float*)d_in[3];
  const float* qkv_b  = (const float*)d_in[4];
  const float* proj_w = (const float*)d_in[5];
  const float* proj_b = (const float*)d_in[6];
  float* out = (float*)d_out;

  char* base = (char*)d_ws;
  size_t off = 0;
  auto alloc = [&](size_t elems, size_t esz) -> void* {
    void* p = base + off;
    off = (off + elems * esz + 255) & ~(size_t)255;
    return p;
  };
  u16* x_bf  = (u16*)alloc((size_t)B * T * C, 2);
  u16* y_bf  = (u16*)alloc((size_t)B * M * C, 2);
  u16* qw_bf = (u16*)alloc((size_t)3 * C * C, 2);
  u16* pw_bf = (u16*)alloc((size_t)C * C, 2);
  u16* q_x   = (u16*)alloc((size_t)BH * T * D, 2);
  u16* k_x   = (u16*)alloc((size_t)BH * T * D, 2);
  u16* vT_x  = (u16*)alloc((size_t)BH * D * T, 2);
  u16* q_y   = (u16*)alloc((size_t)BH * M * D, 2);
  u16* k_y   = (u16*)alloc((size_t)BH * M * D, 2);
  u16* vT_y  = (u16*)alloc((size_t)BH * D * M, 2);
  u16* S1    = (u16*)alloc((size_t)BH * T * M, 2);  // raw scaled x2y scores
  u16* Gms   = (u16*)alloc((size_t)BH * T * M, 2);  // catt_y2x transposed [s,m]
  u16* P1    = (u16*)alloc((size_t)BH * T * M, 2);  // softmax(S1)
  float* cv1 = (float*)alloc((size_t)B * T * C, 4);
  float* cv2 = (float*)alloc((size_t)B * T * C, 4);
  u16* dif   = (u16*)alloc((size_t)B * T * C, 2);

  cvt_bf16_kernel<<<(B * T * C + 255) / 256, 256, 0, stream>>>(x, x_bf, B * T * C);
  cvt_bf16_kernel<<<(B * M * C + 255) / 256, 256, 0, stream>>>(y, y_bf, B * M * C);
  cvt_bf16_kernel<<<(3 * C * C + 255) / 256, 256, 0, stream>>>(qkv_w, qw_bf, 3 * C * C);
  cvt_bf16_kernel<<<(C * C + 255) / 256, 256, 0, stream>>>(proj_w, pw_bf, C * C);

  qkv_gemm_kernel<<<dim3(B * T / 16, 3 * C / 64), 32, 0, stream>>>(
      x_bf, qw_bf, qkv_b, B * T, T, q_x, k_x, vT_x);
  qkv_gemm_kernel<<<dim3(B * M / 16, 3 * C / 64), 32, 0, stream>>>(
      y_bf, qw_bf, qkv_b, B * M, M, q_y, k_y, vT_y);

  score_gemm_kernel<<<dim3(T / 16, M / 64, BH), 32, 0, stream>>>(q_x, k_y, S1, T, M, scale);
  score_gemm_kernel<<<dim3(T / 16, M / 64, BH), 32, 0, stream>>>(k_x, q_y, Gms, T, M, scale);

  softmax_kernel<<<BH * T, 256, 0, stream>>>(S1, P1, M);
  pv_gemm_kernel<<<dim3(T / 16, 1, BH), 32, 0, stream>>>(P1, vT_y, cv1, T, M, H, C);

  chained_attn_kernel<<<dim3(T / 64, 1, BH), 128, 0, stream>>>(
      S1, Gms, vT_x, mask, scale, cv2, T, M, H, C);

  diff_bf_kernel<<<(B * T * C + 255) / 256, 256, 0, stream>>>(cv1, cv2, dif, B * T * C);
  proj_gemm_kernel<<<dim3(B * T / 16, C / 64), 32, 0, stream>>>(dif, pw_bf, proj_b, out, B * T);

  (void)in_sizes; (void)n_in; (void)out_size; (void)ws_size;
}
